// GAT_DGL_84851373900201
// MI455X (gfx1250) — matrix operand order repaired and checked by reading, not compiled, not run on hardware
//
#include <hip/hip_runtime.h>
#include <hip/hip_bf16.h>
#include <math.h>

#define NN 50000
#define DD 64
#define GG 10
#define CC 10
#define NEG_SLOPE 0.2f

typedef __attribute__((ext_vector_type(16))) __bf16 v16bf;
typedef __attribute__((ext_vector_type(8)))  float  v8f;

union V16Load { int4 i[2]; v16bf v; };

// ---------- helpers ----------
__device__ __forceinline__ float lrelu(float v) { return v > 0.f ? v : NEG_SLOPE * v; }
// monotonic unsigned encoding of float (for atomicMax-based segment max)
__device__ __forceinline__ unsigned ordEnc(float f) {
  unsigned b = __float_as_uint(f);
  return (b & 0x80000000u) ? ~b : (b | 0x80000000u);
}
__device__ __forceinline__ float ordDec(unsigned u) {
  unsigned b = (u & 0x80000000u) ? (u ^ 0x80000000u) : ~u;
  return __uint_as_float(b);
}

// ---------- conversion kernels ----------
__global__ void k_f2bf(const float* __restrict__ in, __hip_bfloat16* __restrict__ o, int n) {
  int i = blockIdx.x * blockDim.x + threadIdx.x;
  if (i < n) o[i] = __float2bfloat16(in[i]);
}

__global__ void k_relu_f2bf(const float* __restrict__ in, __hip_bfloat16* __restrict__ o, int n) {
  int i = blockIdx.x * blockDim.x + threadIdx.x;
  if (i < n) { float v = in[i]; o[i] = __float2bfloat16(v > 0.f ? v : 0.f); }
}

// W [K,M] f32 -> Wt [M,K] bf16 (transposed so B-operand loads are contiguous)
__global__ void k_wT2bf(const float* __restrict__ W, __hip_bfloat16* __restrict__ Wt, int K, int M) {
  int i = blockIdx.x * blockDim.x + threadIdx.x;
  if (i < K * M) { int k = i / M, m = i % M; Wt[m * K + k] = __float2bfloat16(W[i]); }
}

// ---------- WMMA GEMM: Z[NN,M] = X[NN,K](bf16) @ Wt[M,K]^T (bf16), f32 accumulate ----------
// one wave per 16x64 output strip: 4 accumulators, A loaded once per k-step (4x A reuse).
// EXEC all ones, no divergence (NN%16==0, M%64==0).
__global__ __launch_bounds__(32) void k_gemm(const __hip_bfloat16* __restrict__ X,
                                             const __hip_bfloat16* __restrict__ Wt,
                                             float* __restrict__ Z, int K, int M) {
  const int tm = blockIdx.x;          // row tile (NN/16)
  const int tn4 = blockIdx.y;         // 64-wide col strip (M/64)
  const int lane = threadIdx.x;
  const int r = lane & 15;
  const bool hi = lane >= 16;
  // A: lane holds row r; elems 0..15 = K(t*32 + (hi?16:0) + 0..15)
  const __hip_bfloat16* Arow = X + (size_t)(tm * 16 + r) * K + (hi ? 16 : 0);
  // B tile j: lane holds col (tn4*64 + j*16 + r); elems 0..15 = K(t*32 + (hi?16:0) + 0..15)
  const __hip_bfloat16* B0 = Wt + (size_t)(tn4 * 64 + 0 * 16 + r) * K + (hi ? 16 : 0);
  const __hip_bfloat16* B1 = Wt + (size_t)(tn4 * 64 + 1 * 16 + r) * K + (hi ? 16 : 0);
  const __hip_bfloat16* B2 = Wt + (size_t)(tn4 * 64 + 2 * 16 + r) * K + (hi ? 16 : 0);
  const __hip_bfloat16* B3 = Wt + (size_t)(tn4 * 64 + 3 * 16 + r) * K + (hi ? 16 : 0);
  v8f acc0 = {}, acc1 = {}, acc2 = {}, acc3 = {};
  const int nk = K >> 5;
  for (int t = 0; t < nk; ++t) {
    V16Load a, b0, b1, b2, b3;
    a.i[0]  = *(const int4*)(Arow + t * 32);
    a.i[1]  = *(const int4*)(Arow + t * 32 + 8);
    b0.i[0] = *(const int4*)(B0 + t * 32);
    b0.i[1] = *(const int4*)(B0 + t * 32 + 8);
    b1.i[0] = *(const int4*)(B1 + t * 32);
    b1.i[1] = *(const int4*)(B1 + t * 32 + 8);
    b2.i[0] = *(const int4*)(B2 + t * 32);
    b2.i[1] = *(const int4*)(B2 + t * 32 + 8);
    b3.i[0] = *(const int4*)(B3 + t * 32);
    b3.i[1] = *(const int4*)(B3 + t * 32 + 8);
    acc0 = __builtin_amdgcn_wmma_f32_16x16x32_bf16(false, a.v, false, b0.v, (short)0, acc0, false, false);
    acc1 = __builtin_amdgcn_wmma_f32_16x16x32_bf16(false, a.v, false, b1.v, (short)0, acc1, false, false);
    acc2 = __builtin_amdgcn_wmma_f32_16x16x32_bf16(false, a.v, false, b2.v, (short)0, acc2, false, false);
    acc3 = __builtin_amdgcn_wmma_f32_16x16x32_bf16(false, a.v, false, b3.v, (short)0, acc3, false, false);
  }
  // C/D layout: VGPR v -> row v + (hi?8:0), col = lane%16
  float* Zp = Z + (size_t)(tm * 16 + (hi ? 8 : 0)) * M + tn4 * 64 + r;
#pragma unroll
  for (int v = 0; v < 8; ++v) {
    Zp[(size_t)v * M + 0]  = acc0[v];
    Zp[(size_t)v * M + 16] = acc1[v];
    Zp[(size_t)v * M + 32] = acc2[v];
    Zp[(size_t)v * M + 48] = acc3[v];
  }
}

// ---------- attention logits: el/er [NN,H] from z [NN,H,DD] ----------
__global__ void k_elr(const float* __restrict__ z, const float* __restrict__ al,
                      const float* __restrict__ ar, float* __restrict__ el,
                      float* __restrict__ er, int H) {
  int i = blockIdx.x * blockDim.x + threadIdx.x;
  if (i >= NN * H) return;
  int n = i / H, h = i % H;
  const float* zp = z + (size_t)n * H * DD + h * DD;
  const float* ap = al + h * DD;
  const float* bp = ar + h * DD;
  float sl = 0.f, sr = 0.f;
#pragma unroll 8
  for (int d = 0; d < DD; ++d) { float v = zp[d]; sl += v * ap[d]; sr += v * bp[d]; }
  el[i] = sl; er[i] = sr;
}

// ---------- edge pass 1: per-dst max (ordered-uint atomicMax) ----------
__global__ void k_edge_max(const int* __restrict__ src, const int* __restrict__ dst,
                           const float* __restrict__ el, const float* __restrict__ er,
                           unsigned* __restrict__ mmax, int E, int H) {
  int e = blockIdx.x * blockDim.x + threadIdx.x;
  if (e >= E) return;
  int s = src[e], d = dst[e];
  for (int h = 0; h < H; ++h) {
    float v = lrelu(el[s * H + h] + er[d * H + h]);
    atomicMax(&mmax[d * H + h], ordEnc(v));
  }
}

// ---------- edge pass 2: per-dst sum of exp ----------
__global__ void k_edge_sum(const int* __restrict__ src, const int* __restrict__ dst,
                           const float* __restrict__ el, const float* __restrict__ er,
                           const unsigned* __restrict__ mmax, float* __restrict__ den,
                           int E, int H) {
  int e = blockIdx.x * blockDim.x + threadIdx.x;
  if (e >= E) return;
  int s = src[e], d = dst[e];
  for (int h = 0; h < H; ++h) {
    float v = lrelu(el[s * H + h] + er[d * H + h]);
    float ex = __expf(v - ordDec(mmax[d * H + h]));
    atomicAdd(&den[d * H + h], ex);
  }
}

// ---------- edge pass 3: out[dst] += alpha * z[src]; block=256, (256/(H*64)) edges/block ----------
__global__ void k_edge_msg(const int* __restrict__ src, const int* __restrict__ dst,
                           const float* __restrict__ el, const float* __restrict__ er,
                           const unsigned* __restrict__ mmax, const float* __restrict__ den,
                           const float* __restrict__ z, float* __restrict__ out,
                           int E, int H) {
  const int span = H * 64;                    // threads per edge (256 for H=4, 64 for H=1)
  int e = blockIdx.x * (256 / span) + threadIdx.x / span;
  if (e >= E) return;
  int lid = threadIdx.x % span;
  int h = lid >> 6, d = lid & 63;
  int s = src[e], t = dst[e];
  float v = lrelu(el[s * H + h] + er[t * H + h]);
  float coef = __expf(v - ordDec(mmax[t * H + h])) / den[t * H + h];
  atomicAdd(&out[(size_t)t * H * DD + h * DD + d],
            coef * z[(size_t)s * H * DD + h * DD + d]);
}

// ---------- graph mean-pool (LDS partial sums; graph_id is sorted => few global atomics) ----------
__global__ void k_pool(const float* __restrict__ x, const int* __restrict__ gid,
                       float* __restrict__ pool, float* __restrict__ cnt) {
  __shared__ float sacc[GG * DD];
  __shared__ float scnt[GG];
  int tid = threadIdx.x;
  for (int i = tid; i < GG * DD; i += 256) sacc[i] = 0.f;
  if (tid < GG) scnt[tid] = 0.f;
  __syncthreads();
  int base = blockIdx.x * 64;
  int d = tid & 63;
  for (int it = 0; it < 16; ++it) {
    int n = base + it * 4 + (tid >> 6);
    if (n < NN) {
      int g = gid[n];
      float v = x[(size_t)n * DD + d];
      v = v > 0.f ? v : 0.f;   // final ReLU fused here
      atomicAdd(&sacc[g * DD + d], v);
      if (d == 0) atomicAdd(&scnt[g], 1.f);
    }
  }
  __syncthreads();
  for (int i = tid; i < GG * DD; i += 256)
    if (sacc[i] != 0.f) atomicAdd(&pool[i], sacc[i]);
  if (tid < GG && scnt[tid] != 0.f) atomicAdd(&cnt[tid], scnt[tid]);
}

// ---------- MLP head + log_softmax over axis 0 (single block) ----------
__global__ void k_head(const float* __restrict__ pool, const float* __restrict__ cnt,
                       const float* __restrict__ fc1w, const float* __restrict__ fc1b,
                       const float* __restrict__ fc2w, const float* __restrict__ fc2b,
                       float* __restrict__ out) {
  __shared__ float sp[GG * DD];
  __shared__ float st[GG * DD];
  __shared__ float sy[GG * CC];
  int tid = threadIdx.x;
  for (int i = tid; i < GG * DD; i += 256) {
    int g = i / DD;
    sp[i] = pool[i] / fmaxf(cnt[g], 1.f);
  }
  __syncthreads();
  for (int i = tid; i < GG * DD; i += 256) {
    int g = i / DD, j = i % DD;
    float acc = fc1b[j];
    for (int d = 0; d < DD; ++d) acc += sp[g * DD + d] * fc1w[d * DD + j];
    st[i] = acc > 0.f ? acc : (__expf(acc) - 1.f);   // ELU (alpha=1)
  }
  __syncthreads();
  for (int i = tid; i < GG * CC; i += 256) {
    int g = i / CC, c = i % CC;
    float acc = fc2b[c];
    for (int j = 0; j < DD; ++j) acc += st[g * DD + j] * fc2w[j * CC + c];
    sy[i] = acc;
  }
  __syncthreads();
  if (tid < CC) {                                    // log_softmax over graphs (axis 0)
    float m = -1e30f;
    for (int g = 0; g < GG; ++g) m = fmaxf(m, sy[g * CC + tid]);
    float s = 0.f;
    for (int g = 0; g < GG; ++g) s += __expf(sy[g * CC + tid] - m);
    float l = m + __logf(s);
    for (int g = 0; g < GG; ++g) out[g * CC + tid] = sy[g * CC + tid] - l;
  }
}

extern "C" void kernel_launch(void* const* d_in, const int* in_sizes, int n_in,
                              void* d_out, int out_size, void* d_ws, size_t ws_size,
                              hipStream_t stream) {
  const float* h   = (const float*)d_in[0];
  const int*   src = (const int*)d_in[1];
  const int*   dst = (const int*)d_in[2];
  const int*   gid = (const int*)d_in[3];
  const float* W[4]  = { (const float*)d_in[4],  (const float*)d_in[7],
                         (const float*)d_in[10], (const float*)d_in[13] };
  const float* al[4] = { (const float*)d_in[5],  (const float*)d_in[8],
                         (const float*)d_in[11], (const float*)d_in[14] };
  const float* ar[4] = { (const float*)d_in[6],  (const float*)d_in[9],
                         (const float*)d_in[12], (const float*)d_in[15] };
  const float* fc1w = (const float*)d_in[16];
  const float* fc1b = (const float*)d_in[17];
  const float* fc2w = (const float*)d_in[18];
  const float* fc2b = (const float*)d_in[19];
  const int E = in_sizes[1];

  // workspace carve-up (256B aligned)
  char* ws = (char*)d_ws;
  size_t off = 0;
  auto take = [&](size_t bytes) -> void* {
    void* p = ws + off;
    off += (bytes + 255) & ~(size_t)255;
    return p;
  };
  __hip_bfloat16* xb  = (__hip_bfloat16*)take((size_t)NN * 256 * 2);
  float*          z   = (float*)take((size_t)NN * 256 * 4);
  float*          ob  = (float*)take((size_t)NN * 256 * 4);
  __hip_bfloat16* wt  = (__hip_bfloat16*)take(256 * 256 * 2);
  float*          el  = (float*)take((size_t)NN * 4 * 4);
  float*          er  = (float*)take((size_t)NN * 4 * 4);
  unsigned*       mmx = (unsigned*)take((size_t)NN * 4 * 4);
  float*          den = (float*)take((size_t)NN * 4 * 4);
  float*          pl  = (float*)take(GG * DD * 4);
  float*          ct  = (float*)take(GG * 4);

  const int Kd[4] = {128, 256, 256, 256};
  const int Md[4] = {256, 256, 256, 64};
  const int Hd[4] = {4, 4, 4, 1};

  // layer-0 input to bf16
  k_f2bf<<<(NN * 128 + 255) / 256, 256, 0, stream>>>(h, xb, NN * 128);

  for (int L = 0; L < 4; ++L) {
    const int K = Kd[L], M = Md[L], H = Hd[L];
    k_wT2bf<<<(K * M + 255) / 256, 256, 0, stream>>>(W[L], wt, K, M);
    dim3 g(NN / 16, M / 64);
    k_gemm<<<g, 32, 0, stream>>>(xb, wt, z, K, M);
    k_elr<<<(NN * H + 255) / 256, 256, 0, stream>>>(z, al[L], ar[L], el, er, H);
    hipMemsetAsync(mmx, 0, (size_t)NN * H * 4, stream);   // 0 == ordEnc(below -inf)
    hipMemsetAsync(den, 0, (size_t)NN * H * 4, stream);
    hipMemsetAsync(ob, 0, (size_t)NN * M * 4, stream);
    k_edge_max<<<(E + 255) / 256, 256, 0, stream>>>(src, dst, el, er, mmx, E, H);
    k_edge_sum<<<(E + 255) / 256, 256, 0, stream>>>(src, dst, el, er, mmx, den, E, H);
    const int epb = 256 / (H * 64);                       // edges per 256-thread block
    k_edge_msg<<<(E + epb - 1) / epb, 256, 0, stream>>>(src, dst, el, er, mmx, den, z, ob, E, H);
    if (L < 3)
      k_relu_f2bf<<<(NN * M + 255) / 256, 256, 0, stream>>>(ob, xb, NN * M);
  }

  hipMemsetAsync(pl, 0, GG * DD * 4, stream);
  hipMemsetAsync(ct, 0, GG * 4, stream);
  k_pool<<<(NN + 63) / 64, 256, 0, stream>>>(ob, gid, pl, ct);
  k_head<<<1, 256, 0, stream>>>(pl, ct, fc1w, fc1b, fc2w, fc2b, (float*)d_out);
}
